// DAGNN_14791867368185
// MI455X (gfx1250) — matrix operand order repaired and checked by reading, not compiled, not run on hardware
//
#include <hip/hip_runtime.h>
#include <hip/hip_bf16.h>
#include <math.h>

#define N_NODES 100000
#define IN_DIMS 512
#define HID_DIM 256
#define C_DIM   64
#define E_EDGES 1600000
#define K_HOPS  10

typedef __attribute__((ext_vector_type(16))) __bf16 v16bf;
typedef __attribute__((ext_vector_type(8)))  float  v8f;

__device__ __forceinline__ float wave_sum32(float v) {
#pragma unroll
  for (int off = 16; off > 0; off >>= 1) v += __shfl_xor(v, off, 32);
  return v;
}
__device__ __forceinline__ float wave_max32(float v) {
#pragma unroll
  for (int off = 16; off > 0; off >>= 1) v = fmaxf(v, __shfl_xor(v, off, 32));
  return v;
}
__device__ __forceinline__ void atomAddF(float* p, float v) {
  __hip_atomic_fetch_add(p, v, __ATOMIC_RELAXED, __HIP_MEMORY_SCOPE_AGENT);
}

// A-fragment (16x32 bf16, wave32): lane L holds row (L&15); with h=L>>4 the
// v16bf element e maps to K = 8*h + (e<8 ? e : 8+e)  -> two runs of 8
// contiguous floats at +8h and +16+8h.
__device__ __forceinline__ v16bf load_a_frag(const float* __restrict__ p) {
  float4 a0 = *(const float4*)(p);
  float4 a1 = *(const float4*)(p + 4);
  float4 a2 = *(const float4*)(p + 16);
  float4 a3 = *(const float4*)(p + 20);
  v16bf r;
  r[0]=(__bf16)a0.x; r[1]=(__bf16)a0.y; r[2]=(__bf16)a0.z; r[3]=(__bf16)a0.w;
  r[4]=(__bf16)a1.x; r[5]=(__bf16)a1.y; r[6]=(__bf16)a1.z; r[7]=(__bf16)a1.w;
  r[8]=(__bf16)a2.x; r[9]=(__bf16)a2.y; r[10]=(__bf16)a2.z; r[11]=(__bf16)a2.w;
  r[12]=(__bf16)a3.x; r[13]=(__bf16)a3.y; r[14]=(__bf16)a3.z; r[15]=(__bf16)a3.w;
  return r;
}

// Pack W [Kdim, ntiles*16] fp32 (row-major) into fragment-native bf16:
// element index = ((kc*ntiles + nt)*32 + lane)*16 + e,
// value = W[kc*32 + 8*(lane>>4) + (e<8 ? e : 8+e), nt*16 + (lane&15)]  (B 32x16 layout).
__global__ void prep_wbf(const float* __restrict__ W, __bf16* __restrict__ Wbf,
                         int ntiles, int total) {
  int idx = blockIdx.x * blockDim.x + threadIdx.x;
  if (idx >= total) return;
  int e    = idx & 15;
  int lane = (idx >> 4) & 31;
  int t    = idx >> 9;
  int nt   = t % ntiles;
  int kc   = t / ntiles;
  int k = kc * 32 + ((lane >> 4) << 3) + (e < 8 ? e : 8 + e);
  int n = (nt << 4) + (lane & 15);
  Wbf[idx] = (__bf16)W[(size_t)k * (size_t)(ntiles << 4) + n];
}

// h1 = relu(x @ W1 + b1); block = 4 waves, 16 rows x 256 cols per block,
// each wave owns 4 16x16 accumulators (reuses its A fragment 4x).
__global__ __launch_bounds__(128) void gemm1_relu(
    const float* __restrict__ x, const __bf16* __restrict__ w1bf,
    const float* __restrict__ b1, float* __restrict__ h1) {
  int lane = threadIdx.x & 31;
  int wave = threadIdx.x >> 5;
  int h = lane >> 4, r15 = lane & 15;
  int tileM = blockIdx.x << 4;
  const float* arow = x + (size_t)(tileM + r15) * IN_DIMS + (h << 3);
  int ntBase = wave << 2;
  v8f acc[4] = {};
  for (int kc = 0; kc < IN_DIMS / 32; ++kc) {
    v16bf a = load_a_frag(arow + kc * 32);
    const __bf16* bb =
        w1bf + ((size_t)(kc * (HID_DIM / 16) + ntBase) << 9) + (lane << 4);
#pragma unroll
    for (int t = 0; t < 4; ++t) {
      v16bf b = *(const v16bf*)(bb + ((size_t)t << 9));
      acc[t] = __builtin_amdgcn_wmma_f32_16x16x32_bf16(
          false, a, false, b, (short)0, acc[t], false, false);
    }
  }
  int rbase = tileM + (h << 3);
#pragma unroll
  for (int t = 0; t < 4; ++t) {
    int col = ((ntBase + t) << 4) + r15;
    float bias = b1[col];
#pragma unroll
    for (int v = 0; v < 8; ++v) {
      float o = acc[t][v] + bias;
      h1[(size_t)(rbase + v) * HID_DIM + col] = o > 0.f ? o : 0.f;
    }
  }
}

// h0 = h1 @ W2 + b2 ; block = 4 waves, 16 rows x 64 cols, 1 acc per wave.
__global__ __launch_bounds__(128) void gemm2(
    const float* __restrict__ h1, const __bf16* __restrict__ w2bf,
    const float* __restrict__ b2, float* __restrict__ hout) {
  int lane = threadIdx.x & 31;
  int wave = threadIdx.x >> 5;  // == column tile
  int h = lane >> 4, r15 = lane & 15;
  int tileM = blockIdx.x << 4;
  const float* arow = h1 + (size_t)(tileM + r15) * HID_DIM + (h << 3);
  v8f acc = {};
  for (int kc = 0; kc < HID_DIM / 32; ++kc) {
    v16bf a = load_a_frag(arow + kc * 32);
    const __bf16* bb =
        w2bf + ((size_t)(kc * (C_DIM / 16) + wave) << 9) + (lane << 4);
    v16bf b = *(const v16bf*)bb;
    acc = __builtin_amdgcn_wmma_f32_16x16x32_bf16(
        false, a, false, b, (short)0, acc, false, false);
  }
  int col = (wave << 4) + r15;
  float bias = b2[col];
  int rbase = tileM + (h << 3);
#pragma unroll
  for (int v = 0; v < 8; ++v)
    hout[(size_t)(rbase + v) * C_DIM + col] = acc[v] + bias;
}

__global__ void deg_init(float* __restrict__ deg) {
  int n = blockIdx.x * blockDim.x + threadIdx.x;
  if (n < N_NODES) deg[n] = 1.0f;  // appended self loop
}

__global__ void deg_edges(const int* __restrict__ ei, float* __restrict__ deg) {
  int e = blockIdx.x * blockDim.x + threadIdx.x;
  if (e >= E_EDGES) return;
  int r = ei[e], c = ei[E_EDGES + e];
  if (r != c) atomAddF(&deg[c], 1.0f);  // original self loops get weight 0
}

__global__ void make_dinv(const float* __restrict__ deg, float* __restrict__ dinv) {
  int n = blockIdx.x * blockDim.x + threadIdx.x;
  if (n < N_NODES) {
    float d = deg[n];
    dinv[n] = d > 0.f ? rsqrtf(d) : 0.f;
  }
}

// emb = sigmoid(h0.Wp + bp) * h0 ; seed = dinv^2 * h0. One wave per node.
__global__ void init_emb_seed(const float* __restrict__ h0,
                              const float* __restrict__ Wp, const float* __restrict__ bp,
                              const float* __restrict__ dinv,
                              float* __restrict__ emb, float* __restrict__ seed) {
  int gid = blockIdx.x * blockDim.x + threadIdx.x;
  int n = gid >> 5, lane = gid & 31;
  if (n >= N_NODES) return;
  float2 hv = *(const float2*)(h0 + (size_t)n * C_DIM + lane * 2);
  float2 wp = *(const float2*)(Wp + lane * 2);
  float dot = wave_sum32(hv.x * wp.x + hv.y * wp.y);
  float s = 1.0f / (1.0f + __expf(-(dot + bp[0])));
  float2 o; o.x = s * hv.x; o.y = s * hv.y;
  *(float2*)(emb + (size_t)n * C_DIM + lane * 2) = o;
  float d2 = dinv[n] * dinv[n];
  float2 sd; sd.x = d2 * hv.x; sd.y = d2 * hv.y;
  *(float2*)(seed + (size_t)n * C_DIM + lane * 2) = sd;
}

// One hop: nxt (pre-seeded with dinv^2*cur) += sum over edges of
// dinv[r]*dinv[c]*cur[r]. 16 lanes per edge, float4 per lane (coalesced).
__global__ void scatter_edges(const int* __restrict__ ei,
                              const float* __restrict__ dinv,
                              const float* __restrict__ cur,
                              float* __restrict__ nxt) {
  long long tid = (long long)blockIdx.x * blockDim.x + threadIdx.x;
  if (tid >= (long long)E_EDGES * 16) return;
  int e = (int)(tid >> 4);
  int g = (int)(tid & 15);
  int r = ei[e], c = ei[E_EDGES + e];
  if (r == c) return;
  float w = dinv[r] * dinv[c];
  float4 v = *(const float4*)(cur + (size_t)r * C_DIM + (g << 2));
  float* dst = nxt + (size_t)c * C_DIM + (g << 2);
  atomAddF(dst + 0, w * v.x);
  atomAddF(dst + 1, w * v.y);
  atomAddF(dst + 2, w * v.z);
  atomAddF(dst + 3, w * v.w);
}

// emb += sigmoid(hk.Wp + bp) * hk ; optionally seed next hop buffer.
__global__ void combine_seed(const float* __restrict__ hk,
                             const float* __restrict__ Wp, const float* __restrict__ bp,
                             const float* __restrict__ dinv,
                             float* __restrict__ emb, float* __restrict__ seed,
                             int doSeed) {
  int gid = blockIdx.x * blockDim.x + threadIdx.x;
  int n = gid >> 5, lane = gid & 31;
  if (n >= N_NODES) return;
  float2 hv = *(const float2*)(hk + (size_t)n * C_DIM + lane * 2);
  float2 wp = *(const float2*)(Wp + lane * 2);
  float dot = wave_sum32(hv.x * wp.x + hv.y * wp.y);
  float s = 1.0f / (1.0f + __expf(-(dot + bp[0])));
  float2 eo = *(float2*)(emb + (size_t)n * C_DIM + lane * 2);
  eo.x += s * hv.x; eo.y += s * hv.y;
  *(float2*)(emb + (size_t)n * C_DIM + lane * 2) = eo;
  if (doSeed) {
    float d2 = dinv[n] * dinv[n];
    float2 sd; sd.x = d2 * hv.x; sd.y = d2 * hv.y;
    *(float2*)(seed + (size_t)n * C_DIM + lane * 2) = sd;
  }
}

__global__ void log_softmax_out(const float* __restrict__ emb,
                                float* __restrict__ outp) {
  int gid = blockIdx.x * blockDim.x + threadIdx.x;
  int n = gid >> 5, lane = gid & 31;
  if (n >= N_NODES) return;
  float2 v = *(const float2*)(emb + (size_t)n * C_DIM + lane * 2);
  float m = wave_max32(fmaxf(v.x, v.y));
  float se = wave_sum32(__expf(v.x - m) + __expf(v.y - m));
  float lse = m + __logf(se);
  float2 o; o.x = v.x - lse; o.y = v.y - lse;
  *(float2*)(outp + (size_t)n * C_DIM + lane * 2) = o;
}

extern "C" void kernel_launch(void* const* d_in, const int* in_sizes, int n_in,
                              void* d_out, int out_size, void* d_ws, size_t ws_size,
                              hipStream_t stream) {
  (void)in_sizes; (void)n_in; (void)out_size; (void)ws_size;
  const float* x  = (const float*)d_in[0];
  const int*   ei = (const int*)d_in[1];
  // d_in[2] is K (known: 10)
  const float* W1 = (const float*)d_in[3];
  const float* b1 = (const float*)d_in[4];
  const float* W2 = (const float*)d_in[5];
  const float* b2 = (const float*)d_in[6];
  const float* Wp = (const float*)d_in[7];
  const float* bp = (const float*)d_in[8];

  char* ws = (char*)d_ws;
  size_t off = 0;
  auto alloc = [&](size_t bytes) -> void* {
    void* p = ws + off;
    off += (bytes + 255) & ~(size_t)255;
    return p;
  };
  float*  h1   = (float*)alloc((size_t)N_NODES * HID_DIM * 4);
  float*  hA   = (float*)alloc((size_t)N_NODES * C_DIM * 4);
  float*  hB   = (float*)alloc((size_t)N_NODES * C_DIM * 4);
  float*  deg  = (float*)alloc((size_t)N_NODES * 4);
  float*  dinv = (float*)alloc((size_t)N_NODES * 4);
  __bf16* w1bf = (__bf16*)alloc((size_t)IN_DIMS * HID_DIM * 2);
  __bf16* w2bf = (__bf16*)alloc((size_t)HID_DIM * C_DIM * 2);

  float* outp = (float*)d_out;                       // log_softmax [N,C]
  float* emb  = outp + (size_t)N_NODES * C_DIM;      // emb accumulator [N,C]

  // weight repack (fragment-native bf16)
  prep_wbf<<<(IN_DIMS * HID_DIM + 255) / 256, 256, 0, stream>>>(
      W1, w1bf, HID_DIM / 16, IN_DIMS * HID_DIM);
  prep_wbf<<<(HID_DIM * C_DIM + 255) / 256, 256, 0, stream>>>(
      W2, w2bf, C_DIM / 16, HID_DIM * C_DIM);

  // MLP encoder via WMMA
  gemm1_relu<<<N_NODES / 16, 128, 0, stream>>>(x, w1bf, b1, h1);
  gemm2<<<N_NODES / 16, 128, 0, stream>>>(h1, w2bf, b2, hA);

  // gcn_norm degrees
  deg_init<<<(N_NODES + 255) / 256, 256, 0, stream>>>(deg);
  deg_edges<<<(E_EDGES + 255) / 256, 256, 0, stream>>>(ei, deg);
  make_dinv<<<(N_NODES + 255) / 256, 256, 0, stream>>>(deg, dinv);

  // k = 0 term + seed first hop
  const int nodeWaveGrid = (N_NODES * 32 + 255) / 256;
  init_emb_seed<<<nodeWaveGrid, 256, 0, stream>>>(hA, Wp, bp, dinv, emb, hB);

  float* cur = hA;
  float* nxt = hB;
  for (int k = 0; k < K_HOPS; ++k) {
    scatter_edges<<<(E_EDGES * 16) / 256, 256, 0, stream>>>(ei, dinv, cur, nxt);
    combine_seed<<<nodeWaveGrid, 256, 0, stream>>>(
        nxt, Wp, bp, dinv, emb, cur, (k < K_HOPS - 1) ? 1 : 0);
    float* t = cur; cur = nxt; nxt = t;
  }

  log_softmax_out<<<nodeWaveGrid, 256, 0, stream>>>(emb, outp);
}
